// MultiHeadAttention_36636071035002
// MI455X (gfx1250) — compile-verified, matrix operand order checked by reading, not run on hardware
//
#include <hip/hip_runtime.h>

#define B_  2
#define S_  2048
#define E_  1024
#define H_  16
#define D_  64
#define HD_ 1024
#define BS_ 4096
#define SCALE_ 0.125f
#define NEGV_ -1e9f

typedef __bf16 bf16;
typedef bf16  v16bf __attribute__((ext_vector_type(16)));
typedef bf16  v8bf  __attribute__((ext_vector_type(8)));
typedef float v8f   __attribute__((ext_vector_type(8)));
typedef unsigned int v4u __attribute__((ext_vector_type(4)));
typedef int v4i __attribute__((ext_vector_type(4)));
typedef int v8i __attribute__((ext_vector_type(8)));

__device__ __forceinline__ v8f wmma_bf16(v16bf a, v16bf b, v8f c) {
  return __builtin_amdgcn_wmma_f32_16x16x32_bf16(false, a, false, b, (short)0, c,
                                                 false, false);
}

// ---- Tensor Data Mover: DMA a [rows x 64] bf16 tile (row stride K elems) to LDS.
// D# per CDNA5 ISA 8.3/8.4: group0 = {count, lds_addr, global_addr, type=2},
// group1 = {data_size=2B, tensor_dim0=K, tensor_dim1=1024, tile_dim0=64,
//           tile_dim1=rows, tensor_dim0_stride=K}. LDS result is row-major [rows][64].
__device__ __forceinline__ void tdm_load_tile(const bf16* gsrc, unsigned ldsOff,
                                              int K, int rows) {
  unsigned long long ga = (unsigned long long)(uintptr_t)gsrc;
  v4u g0;
  g0[0] = 1u;                                                  // count=1, user mode
  g0[1] = ldsOff;                                              // lds_addr (bytes)
  g0[2] = (unsigned)ga;                                        // global_addr[31:0]
  g0[3] = (unsigned)((ga >> 32) & 0x01FFFFFFu) | (2u << 30);   // addr[56:32] | type=2
  v8i g1;
  g1[0] = (int)(1u << 16);                                     // data_size=1 (2 bytes)
  g1[1] = (int)(((unsigned)K & 0xFFFFu) << 16);                // tensor_dim0 lo16
  g1[2] = (int)(((unsigned)K >> 16) | (1024u << 16));          // dim0 hi16 | tensor_dim1 lo16
  g1[3] = (int)(64u << 16);                                    // tensor_dim1 hi16(0) | tile_dim0=64
  g1[4] = (int)(unsigned)rows;                                 // tile_dim1=rows, tile_dim2=0
  g1[5] = (int)(unsigned)K;                                    // tensor_dim0_stride lo32
  g1[6] = 0;                                                   // stride hi | dim1_stride lo
  g1[7] = 0;
  v4i zz4 = {0, 0, 0, 0};
  v8i zz8 = {0, 0, 0, 0, 0, 0, 0, 0};
  __builtin_amdgcn_tensor_load_to_lds(g0, g1, zz4, zz4, zz8, 0);
}

// 16x32 bf16 fragment load from global (A layout; same per-lane K layout works
// for B when stored transposed [N,K] row-major).
__device__ __forceinline__ v16bf load_frag(const bf16* src, int row0, int ld, int kbase) {
  int lane = threadIdx.x & 31;
  int idx  = lane & 15;
  int half = lane >> 4;
  const bf16* p = src + (size_t)(row0 + idx) * ld + kbase + half * 8;
  v8bf lo = *(const v8bf*)p;
  v8bf hi = *(const v8bf*)(p + 16);
  v16bf f;
#pragma unroll
  for (int i = 0; i < 8; ++i) { f[i] = lo[i]; f[i + 8] = hi[i]; }
  return f;
}

// Fragment from an LDS-resident [64][64] bf16 tile.
__device__ __forceinline__ v16bf load_frag_lds(const bf16* tile, int row0, int kbase) {
  int lane = threadIdx.x & 31;
  int idx  = lane & 15;
  int half = lane >> 4;
  const bf16* p = tile + (row0 + idx) * 64 + kbase + half * 8;
  v8bf lo = *(const v8bf*)p;
  v8bf hi = *(const v8bf*)(p + 16);
  v16bf f;
#pragma unroll
  for (int i = 0; i < 8; ++i) { f[i] = lo[i]; f[i + 8] = hi[i]; }
  return f;
}

// Fragment sourced from f32 (prob matrix), converted to bf16 on the fly.
__device__ __forceinline__ v16bf load_frag_f32(const float* src, int row0, int ld, int kbase) {
  int lane = threadIdx.x & 31;
  int idx  = lane & 15;
  int half = lane >> 4;
  const float* p = src + (size_t)(row0 + idx) * ld + kbase + half * 8;
  v16bf f;
#pragma unroll
  for (int i = 0; i < 8; ++i) { f[i] = (bf16)p[i]; f[i + 8] = (bf16)p[i + 16]; }
  return f;
}

// ---------------- prep kernels ----------------

__global__ void __launch_bounds__(256) f32_to_bf16_kernel(const float* x, bf16* o, int n) {
  int i = blockIdx.x * 256 + threadIdx.x;
  if (i < n) o[i] = (bf16)x[i];
}

// w: [K,N] f32 row-major -> wt: [N,K] bf16 row-major
__global__ void __launch_bounds__(256) transpose_bf16_kernel(const float* w, bf16* wt,
                                                             int K, int N) {
  int e = blockIdx.x * 256 + threadIdx.x;
  if (e >= K * N) return;
  int k = e / N, n = e - k * N;
  wt[(size_t)n * K + k] = (bf16)w[e];
}

// ------------- GEMM + bias (WMMA; TDM-staged B tiles in LDS) -------------
// C[M,N] = A[M,K](bf16) * Bt[N,K](bf16)^T + bias
// B tile (64 rows x 64 K) is DMA'd into LDS by the Tensor Data Mover (wave 0),
// double-buffered; all 8 waves read fragments from LDS. A stays in registers,
// ping-pong double-buffered from global.
// MODE 0: f32 row-major [M,N] store; MODE 1: bf16 [B,H,S,D]; MODE 2: bf16 [B,H,D,S]
template <int K, int MODE>
__global__ void __launch_bounds__(256) gemm_bias_kernel(const bf16* A, const bf16* Bt,
                                                        const float* bias, void* outp,
                                                        int N) {
  __shared__ bf16 btile[2][64 * 64];  // 2 x 8 KB
  int wave = threadIdx.x >> 5;
  int m0 = blockIdx.x * 128 + wave * 16;
  int n0 = blockIdx.y * 64;
  v8f acc[4] = {v8f{}, v8f{}, v8f{}, v8f{}};

  if (wave == 0)
    tdm_load_tile(Bt + (size_t)n0 * K, (unsigned)(uintptr_t)&btile[0][0], K, 64);

  v16bf ax = load_frag(A, m0, K, 0);
  v16bf ay = load_frag(A, m0, K, 32);

  if (wave == 0) __builtin_amdgcn_s_wait_tensorcnt(0);
  __syncthreads();

  int buf = 0;
  for (int kb = 0; kb < K; kb += 64) {
    if (wave == 0 && kb + 64 < K)
      tdm_load_tile(Bt + (size_t)n0 * K + (kb + 64),
                    (unsigned)(uintptr_t)&btile[buf ^ 1][0], K, 64);
    const bf16* tp = &btile[buf][0];
#pragma unroll
    for (int j = 0; j < 4; ++j) acc[j] = wmma_bf16(ax, load_frag_lds(tp, j * 16, 0), acc[j]);
    if (kb + 64 < K) ax = load_frag(A, m0, K, kb + 64);
#pragma unroll
    for (int j = 0; j < 4; ++j) acc[j] = wmma_bf16(ay, load_frag_lds(tp, j * 16, 32), acc[j]);
    if (kb + 64 < K) ay = load_frag(A, m0, K, kb + 96);
    if (wave == 0) __builtin_amdgcn_s_wait_tensorcnt(0);
    __syncthreads();
    buf ^= 1;
  }

  int lane = threadIdx.x & 31, idx = lane & 15, half = lane >> 4;
#pragma unroll
  for (int j = 0; j < 4; ++j) {
#pragma unroll
    for (int r = 0; r < 8; ++r) {
      int m = m0 + half * 8 + r;
      int n = n0 + j * 16 + idx;
      float v = acc[j][r] + bias[n];
      if (MODE == 0) {
        ((float*)outp)[(size_t)m * N + n] = v;
      } else {
        int bb = m / S_, s = m & (S_ - 1);
        int h = n / D_, d = n & (D_ - 1);
        bf16* o = (bf16*)outp;
        if (MODE == 1)
          o[(((size_t)(bb * H_ + h)) * S_ + s) * D_ + d] = (bf16)v;
        else
          o[(((size_t)(bb * H_ + h)) * D_ + d) * S_ + s] = (bf16)v;
      }
    }
  }
}

// ---------------- fused scores + softmax ----------------
// Block = 16 query rows x all 2048 keys for one (b,h).
// Scores (WMMA, scale+mask fused) -> 128 KB LDS tile -> row softmax -> single
// normalized prob write to HBM.
__global__ void __launch_bounds__(256) attn_softmax_kernel(const bf16* q, const bf16* k,
                                                           const unsigned char* mask,
                                                           float* prob) {
  extern __shared__ float sc[];  // [16][S_] = 128 KB
  int bh = blockIdx.y;
  int b = bh >> 4;  // / H_
  int sq0 = blockIdx.x * 16;
  int wave = threadIdx.x >> 5, lane = threadIdx.x & 31;
  int idx = lane & 15, half = lane >> 4;
  const bf16* qh = q + (size_t)bh * S_ * D_;
  const bf16* kh = k + (size_t)bh * S_ * D_;
  const unsigned char* mrow = mask + (size_t)b * S_ * S_;

  v16bf a0 = load_frag(qh, sq0, D_, 0);
  v16bf a1 = load_frag(qh, sq0, D_, 32);

  auto emit = [&](v8f acc, int kt) {
    int sk = kt * 16 + idx;
#pragma unroll
    for (int r = 0; r < 8; ++r) {
      int mq = half * 8 + r;
      float v = acc[r] * SCALE_;
      if (mrow[(size_t)(sq0 + mq) * S_ + sk]) v = NEGV_;
      sc[mq * S_ + sk] = v;
    }
  };

  // each wave streams 16 key tiles (kt = wave + 8*t); ping-pong buffers X/Y
  const int NT = S_ / 16 / 8;  // 16 tiles per wave
  v16bf bx0 = load_frag(kh, (wave + 0 * 8) * 16, D_, 0);
  v16bf bx1 = load_frag(kh, (wave + 0 * 8) * 16, D_, 32);
  v16bf by0 = load_frag(kh, (wave + 1 * 8) * 16, D_, 0);
  v16bf by1 = load_frag(kh, (wave + 1 * 8) * 16, D_, 32);
  for (int t = 0; t < NT - 2; t += 2) {
    {
      v8f acc = v8f{};
      acc = wmma_bf16(a0, bx0, acc);
      acc = wmma_bf16(a1, bx1, acc);
      emit(acc, wave + t * 8);
    }
    bx0 = load_frag(kh, (wave + (t + 2) * 8) * 16, D_, 0);
    bx1 = load_frag(kh, (wave + (t + 2) * 8) * 16, D_, 32);
    {
      v8f acc = v8f{};
      acc = wmma_bf16(a0, by0, acc);
      acc = wmma_bf16(a1, by1, acc);
      emit(acc, wave + (t + 1) * 8);
    }
    by0 = load_frag(kh, (wave + (t + 3) * 8) * 16, D_, 0);
    by1 = load_frag(kh, (wave + (t + 3) * 8) * 16, D_, 32);
  }
  {
    v8f acc = v8f{};
    acc = wmma_bf16(a0, bx0, acc);
    acc = wmma_bf16(a1, bx1, acc);
    emit(acc, wave + (NT - 2) * 8);
  }
  {
    v8f acc = v8f{};
    acc = wmma_bf16(a0, by0, acc);
    acc = wmma_bf16(a1, by1, acc);
    emit(acc, wave + (NT - 1) * 8);
  }
  __syncthreads();

  // softmax: wave w handles rows w and w+8; wave-level reduction via shfl_xor
#pragma unroll
  for (int rr = 0; rr < 2; ++rr) {
    int r = wave + rr * 8;
    float* row = sc + r * S_;
    float mx = -3.4e38f;
    for (int i = lane; i < S_; i += 32) mx = fmaxf(mx, row[i]);
#pragma unroll
    for (int off = 16; off; off >>= 1) mx = fmaxf(mx, __shfl_xor(mx, off));
    float sum = 0.f;
    for (int i = lane; i < S_; i += 32) {
      float e = __expf(row[i] - mx);
      row[i] = e;
      sum += e;
    }
#pragma unroll
    for (int off = 16; off; off >>= 1) sum += __shfl_xor(sum, off);
    float inv = 1.0f / sum;
    float* pr = prob + ((size_t)bh * S_ + sq0 + r) * S_;
    for (int i = lane; i < S_; i += 32) pr[i] = row[i] * inv;
  }
}

// ---------------- context = prob @ v (WMMA, ping-pong pipeline) ----------------
// vT stored [B,H,D,S]; ctx written bf16 [B,S,H*D]
__global__ void __launch_bounds__(256) context_kernel(const float* prob, const bf16* vT,
                                                      bf16* ctx) {
  int bh = blockIdx.y;
  int b = bh >> 4, h = bh & (H_ - 1);
  int wave = threadIdx.x >> 5;
  int mw = wave >> 2, nw = wave & 3;
  int m0 = blockIdx.x * 32 + mw * 16;
  int n0 = nw * 16;  // d offset
  const float* ph = prob + (size_t)bh * S_ * S_;
  const bf16* vh = vT + (size_t)bh * D_ * S_;
  v8f acc = v8f{};

  v16bf ax = load_frag_f32(ph, m0, S_, 0);
  v16bf bx = load_frag(vh, n0, S_, 0);
  v16bf ay = load_frag_f32(ph, m0, S_, 32);
  v16bf by = load_frag(vh, n0, S_, 32);
  for (int kb = 0; kb < S_ - 64; kb += 64) {
    acc = wmma_bf16(ax, bx, acc);
    ax = load_frag_f32(ph, m0, S_, kb + 64);
    bx = load_frag(vh, n0, S_, kb + 64);
    acc = wmma_bf16(ay, by, acc);
    ay = load_frag_f32(ph, m0, S_, kb + 96);
    by = load_frag(vh, n0, S_, kb + 96);
  }
  acc = wmma_bf16(ax, bx, acc);
  acc = wmma_bf16(ay, by, acc);

  int lane = threadIdx.x & 31, idx = lane & 15, half = lane >> 4;
#pragma unroll
  for (int r = 0; r < 8; ++r) {
    int s = m0 + half * 8 + r;
    int d = n0 + idx;
    ctx[(((size_t)b * S_ + s) * H_ + h) * D_ + d] = (bf16)acc[r];
  }
}

extern "C" void kernel_launch(void* const* d_in, const int* in_sizes, int n_in,
                              void* d_out, int out_size, void* d_ws, size_t ws_size,
                              hipStream_t stream) {
  const float* Q  = (const float*)d_in[0];
  const float* K  = (const float*)d_in[1];
  const float* V  = (const float*)d_in[2];
  const unsigned char* mask = (const unsigned char*)d_in[3];
  const float* WQ = (const float*)d_in[4];
  const float* bQ = (const float*)d_in[5];
  const float* WK = (const float*)d_in[6];
  const float* bK = (const float*)d_in[7];
  const float* WV = (const float*)d_in[8];
  const float* bV = (const float*)d_in[9];
  const float* WO = (const float*)d_in[10];
  const float* bO = (const float*)d_in[11];

  float* y    = (float*)d_out;
  float* prob = y + (size_t)B_ * S_ * E_;

  // workspace layout (bytes)
  char* ws = (char*)d_ws;
  const size_t ACT = (size_t)BS_ * E_ * sizeof(bf16);   // 8 MB
  const size_t WT  = (size_t)E_ * HD_ * sizeof(bf16);   // 2 MB
  bf16* Xq  = (bf16*)(ws + 0 * ACT);
  bf16* Xk  = (bf16*)(ws + 1 * ACT);
  bf16* Xv  = (bf16*)(ws + 2 * ACT);
  bf16* WQt = (bf16*)(ws + 3 * ACT + 0 * WT);
  bf16* WKt = (bf16*)(ws + 3 * ACT + 1 * WT);
  bf16* WVt = (bf16*)(ws + 3 * ACT + 2 * WT);
  bf16* WOt = (bf16*)(ws + 3 * ACT + 3 * WT);
  bf16* qb  = (bf16*)(ws + 3 * ACT + 4 * WT + 0 * ACT);  // [B,H,S,D]
  bf16* kb  = (bf16*)(ws + 3 * ACT + 4 * WT + 1 * ACT);  // [B,H,S,D]
  bf16* vT  = (bf16*)(ws + 3 * ACT + 4 * WT + 2 * ACT);  // [B,H,D,S]
  bf16* ctx = (bf16*)(ws + 3 * ACT + 4 * WT + 3 * ACT);  // [B,S,H*D]

  const int nAct = BS_ * E_;
  dim3 cvtGrid((nAct + 255) / 256);
  f32_to_bf16_kernel<<<cvtGrid, 256, 0, stream>>>(Q, Xq, nAct);
  f32_to_bf16_kernel<<<cvtGrid, 256, 0, stream>>>(K, Xk, nAct);
  f32_to_bf16_kernel<<<cvtGrid, 256, 0, stream>>>(V, Xv, nAct);

  dim3 trGrid((E_ * HD_ + 255) / 256);
  transpose_bf16_kernel<<<trGrid, 256, 0, stream>>>(WQ, WQt, E_, HD_);
  transpose_bf16_kernel<<<trGrid, 256, 0, stream>>>(WK, WKt, E_, HD_);
  transpose_bf16_kernel<<<trGrid, 256, 0, stream>>>(WV, WVt, E_, HD_);
  transpose_bf16_kernel<<<trGrid, 256, 0, stream>>>(WO, WOt, HD_, E_);

  dim3 gemmGrid(BS_ / 128, HD_ / 64);
  gemm_bias_kernel<E_, 1><<<gemmGrid, 256, 0, stream>>>(Xq, WQt, bQ, qb, HD_);
  gemm_bias_kernel<E_, 1><<<gemmGrid, 256, 0, stream>>>(Xk, WKt, bK, kb, HD_);
  gemm_bias_kernel<E_, 2><<<gemmGrid, 256, 0, stream>>>(Xv, WVt, bV, vT, HD_);

  dim3 atGrid(S_ / 16, B_ * H_);
  attn_softmax_kernel<<<atGrid, 256, 16 * S_ * sizeof(float), stream>>>(qb, kb, mask, prob);

  dim3 cxGrid(S_ / 32, B_ * H_);
  context_kernel<<<cxGrid, 256, 0, stream>>>(prob, vT, ctx);

  gemm_bias_kernel<HD_, 0><<<gemmGrid, 256, 0, stream>>>(ctx, WOt, bO, y, E_);
}
